// SparselyGatedMixtureOfExperts_13073880449222
// MI455X (gfx1250) — matrix operand order repaired and checked by reading, not compiled, not run on hardware
//
#include <hip/hip_runtime.h>

#define N_TOKENS  65536
#define D_MODEL   256
#define N_EXPERTS 32
#define TILE_M    64
#define MAX_TILES (N_EXPERTS + (2 * N_TOKENS) / TILE_M)  // 32 + 2048 = 2080

typedef __bf16 bf16_t;
typedef __attribute__((ext_vector_type(8)))  bf16_t v8bf;
typedef __attribute__((ext_vector_type(16))) bf16_t v16bf;
typedef __attribute__((ext_vector_type(8)))  float  v8f;

__device__ __forceinline__ bf16_t f2bf(float f) { return (bf16_t)f; }

// ---------------------------------------------------------------------------
// Zero-fill helpers (harness poisons d_out / d_ws with 0xAA).
// ---------------------------------------------------------------------------
__global__ void zero_f32(float* __restrict__ p, int n4) {
  int i = blockIdx.x * blockDim.x + threadIdx.x;
  if (i < n4) ((float4*)p)[i] = make_float4(0.f, 0.f, 0.f, 0.f);
}
__global__ void zero_i32(int* __restrict__ p, int n) {
  int i = blockIdx.x * blockDim.x + threadIdx.x;
  if (i < n) p[i] = 0;
}

// ---------------------------------------------------------------------------
// 1) Gating: one wave per token, one lane per expert (wave32).
//    H[t,e] = (x Wg)[t,e] + noise_vec[e] * softplus((x Wnoise)[t,e])
//    top-2 by wave shuffle reduction, softmax over the two kept logits.
// ---------------------------------------------------------------------------
__global__ void __launch_bounds__(256)
moe_gate(const float* __restrict__ x, const float* __restrict__ Wg,
         const float* __restrict__ Wn, const float* __restrict__ nvec,
         int2* __restrict__ routeE, float2* __restrict__ routeW,
         int* __restrict__ counts) {
  __shared__ float sx[8][D_MODEL];
  const int tid = threadIdx.x, wv = tid >> 5, l = tid & 31;
  const int t = blockIdx.x * 8 + wv;

  const float* xr = x + (size_t)t * D_MODEL;
#pragma unroll
  for (int i = 0; i < 8; ++i) sx[wv][l + 32 * i] = xr[l + 32 * i];
  __syncthreads();

  float ag = 0.f, an = 0.f;
#pragma unroll 4
  for (int d = 0; d < D_MODEL; ++d) {
    const float xv = sx[wv][d];  // LDS broadcast within the wave
    ag = fmaf(xv, Wg[d * N_EXPERTS + l], ag);
    an = fmaf(xv, Wn[d * N_EXPERTS + l], an);
  }
  const float sp = fmaxf(an, 0.f) + log1pf(expf(-fabsf(an)));  // stable softplus
  const float h  = ag + nvec[l] * sp;

  // top-1 (max value, lowest index on ties)
  float v1 = h; int i1 = l;
#pragma unroll
  for (int off = 16; off > 0; off >>= 1) {
    const float ov = __shfl_xor(v1, off, 32);
    const int   oi = __shfl_xor(i1, off, 32);
    if (ov > v1 || (ov == v1 && oi < i1)) { v1 = ov; i1 = oi; }
  }
  // top-2: exclude i1
  float v2 = (l == i1) ? -3.0e38f : h; int i2 = l;
#pragma unroll
  for (int off = 16; off > 0; off >>= 1) {
    const float ov = __shfl_xor(v2, off, 32);
    const int   oi = __shfl_xor(i2, off, 32);
    if (ov > v2 || (ov == v2 && oi < i2)) { v2 = ov; i2 = oi; }
  }

  if (l == 0) {
    const float w0 = 1.f / (1.f + expf(v2 - v1));  // softmax over {v1,v2}
    routeE[t] = make_int2(i1, i2);
    routeW[t] = make_float2(w0, 1.f - w0);
    atomicAdd(&counts[i1], 1);
    atomicAdd(&counts[i2], 1);
  }
}

// ---------------------------------------------------------------------------
// 2) Tiny scan: entry offsets (exact) + tile offsets (for static grid bound).
// ---------------------------------------------------------------------------
__global__ void moe_scan(const int* __restrict__ counts,
                         int* __restrict__ entryOff, int* __restrict__ tileOff) {
  if (threadIdx.x == 0) {
    int off = 0, toff = 0;
    for (int e = 0; e < N_EXPERTS; ++e) {
      entryOff[e] = off; tileOff[e] = toff;
      off  += counts[e];
      toff += (counts[e] + TILE_M - 1) / TILE_M;
    }
    entryOff[N_EXPERTS] = off; tileOff[N_EXPERTS] = toff;
  }
}

// ---------------------------------------------------------------------------
// 3) Scatter tokens into per-expert compacted lists (total = exactly 2N).
// ---------------------------------------------------------------------------
__global__ void __launch_bounds__(256)
moe_scatter(const int2* __restrict__ routeE, const float2* __restrict__ routeW,
            const int* __restrict__ entryOff, int* __restrict__ cursor,
            int* __restrict__ entryTok, float* __restrict__ entryW) {
  const int t = blockIdx.x * blockDim.x + threadIdx.x;
  if (t >= N_TOKENS) return;
  const int2 ee = routeE[t];
  const float2 ww = routeW[t];
  int p = atomicAdd(&cursor[ee.x], 1);
  entryTok[entryOff[ee.x] + p] = t; entryW[entryOff[ee.x] + p] = ww.x;
  p = atomicAdd(&cursor[ee.y], 1);
  entryTok[entryOff[ee.y] + p] = t; entryW[entryOff[ee.y] + p] = ww.y;
}

// ---------------------------------------------------------------------------
// 4) Grouped expert GEMM: one block = (expert, 64-token tile), 8 waves.
//    Each wave owns a 16(M) x 128(N) C strip: 8 accumulators of v8f.
//    A (gathered x rows) and a 32-wide K-slab of We[e] staged in LDS as bf16;
//    B slab stored transposed so fragments are contiguous ds_load_b128's.
// ---------------------------------------------------------------------------
__global__ void __launch_bounds__(256)
moe_expert_gemm(const float* __restrict__ x, const float* __restrict__ We,
                const float* __restrict__ be,
                const int* __restrict__ entryTok, const float* __restrict__ entryW,
                const int* __restrict__ entryOff, const int* __restrict__ tileOff,
                float* __restrict__ out) {
  __shared__ __align__(16) bf16_t As[TILE_M][D_MODEL + 8];  // 264: stride 528B (33*16)
  __shared__ __align__(16) bf16_t BsT[D_MODEL][56];         // transposed slab, 112B rows
  __shared__ int   stok[TILE_M];
  __shared__ float sw[TILE_M];

  const int tile = blockIdx.x;
  if (tile >= tileOff[N_EXPERTS]) return;  // uniform early-exit for padding blocks
  int e = 0;
  while (tileOff[e + 1] <= tile) ++e;      // <=31 scalar iterations
  const int tIn       = tile - tileOff[e];
  const int base      = entryOff[e] + tIn * TILE_M;
  const int cnt       = entryOff[e + 1] - entryOff[e];
  const int rowsValid = min(TILE_M, cnt - tIn * TILE_M);

  const int tid = threadIdx.x, wv = tid >> 5, l = tid & 31;

  if (tid < TILE_M) {
    const bool v = tid < rowsValid;
    stok[tid] = v ? entryTok[base + tid] : 0;
    sw[tid]   = v ? entryW[base + tid]   : 0.f;  // w==0 rows are masked in epilogue
  }
  __syncthreads();

  // Stage gathered A tile (fp32 -> bf16): 4 threads per row, 64 floats each.
  {
    const int row = tid >> 2;
    const int cb  = (tid & 3) * 64;
    const float* xr = x + (size_t)stok[row] * D_MODEL + cb;
#pragma unroll
    for (int i = 0; i < 64; i += 4) {
      const float4 v = *(const float4*)(xr + i);
      As[row][cb + i + 0] = f2bf(v.x);
      As[row][cb + i + 1] = f2bf(v.y);
      As[row][cb + i + 2] = f2bf(v.z);
      As[row][cb + i + 3] = f2bf(v.w);
    }
  }

  const float* WeE = We + (size_t)e * D_MODEL * D_MODEL;
  const float* beE = be + (size_t)e * D_MODEL;

  const int m16   = (wv >> 1) * 16;   // 4 wave-rows cover M=64
  const int nbase = (wv & 1) * 128;   // 2 wave-cols cover N=256

  const v8f vzero = {0.f, 0.f, 0.f, 0.f, 0.f, 0.f, 0.f, 0.f};
  v8f acc[8];
#pragma unroll
  for (int j = 0; j < 8; ++j) acc[j] = vzero;

  for (int ks = 0; ks < D_MODEL; ks += 32) {
    // Stage B slab transposed: thread = output column n, loop K (coalesced per k).
    {
      const int n = tid;
      const float* wp = WeE + (size_t)ks * D_MODEL + n;
#pragma unroll
      for (int k = 0; k < 32; ++k) BsT[n][k] = f2bf(wp[(size_t)k * D_MODEL]);
    }
    if (ks + 32 < D_MODEL)
      __builtin_prefetch(WeE + (size_t)(ks + 32) * D_MODEL + tid, 0, 0);
    __syncthreads();

    // A fragment (16x32 bf16): lanes 0-15 -> K {0..7,16..23}, 16-31 -> {8..15,24..31}
    const int am = m16 + (l & 15);
    const int ak = ks + ((l < 16) ? 0 : 8);
    const v8bf alo = *(const v8bf*)&As[am][ak];
    const v8bf ahi = *(const v8bf*)&As[am][ak + 16];
    const v16bf a = __builtin_shufflevector(alo, ahi,
        0, 1, 2, 3, 4, 5, 6, 7, 8, 9, 10, 11, 12, 13, 14, 15);

#pragma unroll
    for (int j = 0; j < 8; ++j) {
      // B fragment (32x16 bf16): lane = column, lanes 0-15 K {0..7,16..23}, 16-31 K {8..15,24..31}
      const int n  = nbase + 16 * j + (l & 15);
      const int kb = (l < 16) ? 0 : 8;
      const v8bf blo = *(const v8bf*)&BsT[n][kb];
      const v8bf bhi = *(const v8bf*)&BsT[n][kb + 16];
      const v16bf b = __builtin_shufflevector(blo, bhi,
          0, 1, 2, 3, 4, 5, 6, 7, 8, 9, 10, 11, 12, 13, 14, 15);
      acc[j] = __builtin_amdgcn_wmma_f32_16x16x32_bf16(
          false, a, false, b, (short)0, acc[j], false, false);
    }
    __syncthreads();
  }

  // Epilogue: out[token, n] += w * (acc + be[e, n]); two experts/token -> atomics.
  const int hiHalf = (l < 16) ? 0 : 8;  // C layout: VGPR r -> M = r (+8 for hi lanes)
#pragma unroll
  for (int j = 0; j < 8; ++j) {
    const int nn = nbase + 16 * j + (l & 15);
    const float bev = beE[nn];
#pragma unroll
    for (int r = 0; r < 8; ++r) {
      const int m = m16 + r + hiHalf;
      const float w = sw[m];
      if (w != 0.f)
        unsafeAtomicAdd(&out[(size_t)stok[m] * D_MODEL + nn], w * (acc[j][r] + bev));
    }
  }
}

// ---------------------------------------------------------------------------
// Launch
// ---------------------------------------------------------------------------
extern "C" void kernel_launch(void* const* d_in, const int* in_sizes, int n_in,
                              void* d_out, int out_size, void* d_ws, size_t ws_size,
                              hipStream_t stream) {
  (void)in_sizes; (void)n_in; (void)ws_size;
  const float* x    = (const float*)d_in[0];
  const float* Wg   = (const float*)d_in[1];
  const float* Wn   = (const float*)d_in[2];
  const float* We   = (const float*)d_in[3];
  const float* be   = (const float*)d_in[4];
  const float* nvec = (const float*)d_in[5];
  float* out = (float*)d_out;

  char* ws = (char*)d_ws;                       // layout (16B aligned slots):
  int*    counts   = (int*)(ws + 0);            // 32 ints
  int*    cursor   = (int*)(ws + 128);          // 32 ints
  int*    entryOff = (int*)(ws + 256);          // 33 ints
  int*    tileOff  = (int*)(ws + 512);          // 33 ints
  int2*   routeE   = (int2*)(ws + 1024);
  float2* routeW   = (float2*)(ws + 1024 + (size_t)N_TOKENS * 8);
  int*    entryTok = (int*)(ws + 1024 + (size_t)N_TOKENS * 16);
  float*  entryW   = (float*)(ws + 1024 + (size_t)N_TOKENS * 16 + (size_t)2 * N_TOKENS * 4);

  // Zero atomic counters (counts+cursor contiguous) and the output accumulator.
  zero_i32<<<1, 64, 0, stream>>>(counts, 64);
  const int n4 = out_size / 4;
  zero_f32<<<(n4 + 255) / 256, 256, 0, stream>>>(out, n4);

  moe_gate<<<N_TOKENS / 8, 256, 0, stream>>>(x, Wg, Wn, nvec, routeE, routeW, counts);
  moe_scan<<<1, 32, 0, stream>>>(counts, entryOff, tileOff);
  moe_scatter<<<N_TOKENS / 256, 256, 0, stream>>>(routeE, routeW, entryOff, cursor,
                                                  entryTok, entryW);
  moe_expert_gemm<<<MAX_TILES, 256, 0, stream>>>(x, We, be, entryTok, entryW,
                                                 entryOff, tileOff, out);
}
